// VQVAE_57741540328050
// MI455X (gfx1250) — matrix operand order repaired and checked by reading, not compiled, not run on hardware
//
#include <hip/hip_runtime.h>
#include <hip/hip_bf16.h>

// ---------------------------------------------------------------------------
// VQ-VAE forward for MI455X (gfx1250, wave32, WMMA).
// Internal format: activations NHWC _Float16, weights repacked [O][kh][kw][C]
// _Float16, f32 accumulation in v_wmma_f32_16x16x32_f16. All inter-layer
// tensors (16 MB each at 64x64) are L2-resident; compute-bound on WMMA.
// Conv kernels are templated on shape+flags so tap loops fully unroll and
// no scalar branching remains in the K-loop.
// ---------------------------------------------------------------------------

typedef __attribute__((ext_vector_type(16))) _Float16 v16h;
typedef __attribute__((ext_vector_type(8)))  _Float16 v8h;
typedef __attribute__((ext_vector_type(8)))  float    v8f;

#define FLAG_RELU_IN  1
#define FLAG_RELU_OUT 2
#define FLAG_ADD_RES  4

static __device__ __forceinline__ v8h zero8h() {
  v8h z;
#pragma unroll
  for (int i = 0; i < 8; ++i) z[i] = (_Float16)0.0f;
  return z;
}
static __device__ __forceinline__ v8f zero8f() {
  v8f z;
#pragma unroll
  for (int i = 0; i < 8; ++i) z[i] = 0.0f;
  return z;
}
static __device__ __forceinline__ v8h relu8(v8h a) {
#pragma unroll
  for (int i = 0; i < 8; ++i) a[i] = a[i] > (_Float16)0.0f ? a[i] : (_Float16)0.0f;
  return a;
}
#define CAT16(a, b) __builtin_shufflevector(a, b, 0,1,2,3,4,5,6,7,8,9,10,11,12,13,14,15)

// ---------------------------------------------------------------------------
// Weight repack: OIHW f32 -> [O][kh][kw][I] f16   (normal conv)
// ---------------------------------------------------------------------------
__global__ void repack_w_kernel(const float* __restrict__ w, _Float16* __restrict__ o,
                                int O, int I, int KH, int KW) {
  long t = (long)blockIdx.x * 256 + threadIdx.x;
  long total = (long)O * I * KH * KW;
  if (t >= total) return;
  int i  = (int)(t % I);  long r = t / I;
  int kw = (int)(r % KW); r /= KW;
  int kh = (int)(r % KH);
  int oo = (int)(r / KH);
  o[t] = (_Float16)w[(((long)oo * I + i) * KH + kh) * KW + kw];
}

// ConvTranspose weight (I, O, kH, kW) f32 -> [O][kh][kw][I] f16 (no flip:
// direct formulation y[o,oh,ow] += x[i,ih,iw] * w[i][o][kh][kw]).
__global__ void repack_wt_kernel(const float* __restrict__ w, _Float16* __restrict__ o,
                                 int O, int I, int KH, int KW) {
  long t = (long)blockIdx.x * 256 + threadIdx.x;
  long total = (long)O * I * KH * KW;
  if (t >= total) return;
  int i  = (int)(t % I);  long r = t / I;
  int kw = (int)(r % KW); r /= KW;
  int kh = (int)(r % KH);
  int oo = (int)(r / KH);
  o[t] = (_Float16)w[(((long)i * O + oo) * KH + kh) * KW + kw];
}

__global__ void cvt_f32_to_f16_kernel(const float* __restrict__ a, _Float16* __restrict__ o, long n) {
  long t = (long)blockIdx.x * 256 + threadIdx.x;
  if (t < n) o[t] = (_Float16)a[t];
}

// ---------------------------------------------------------------------------
// Stem conv: x f32 NCHW (16,3,256,256) -> a1 f16 NHWC (16,128,128,128ch), k4 s2 p1 + relu
// ---------------------------------------------------------------------------
__global__ __launch_bounds__(256) void conv1_valu_kernel(
    const float* __restrict__ x, const float* __restrict__ w,
    const float* __restrict__ b, _Float16* __restrict__ out) {
  long t = (long)blockIdx.x * 256 + threadIdx.x;   // 16*128*128*128 outputs
  int c = (int)(t & 127); long pix = t >> 7;
  int ow = (int)(pix & 127);
  int oh = (int)((pix >> 7) & 127);
  int n  = (int)(pix >> 14);
  float s = b[c];
  for (int ci = 0; ci < 3; ++ci) {
#pragma unroll
    for (int kh = 0; kh < 4; ++kh) {
      int ih = oh * 2 - 1 + kh;
      if ((unsigned)ih >= 256u) continue;
#pragma unroll
      for (int kw = 0; kw < 4; ++kw) {
        int iw = ow * 2 - 1 + kw;
        if ((unsigned)iw >= 256u) continue;
        s += x[(((long)n * 3 + ci) * 256 + ih) * 256 + iw] *
             w[(((long)c * 3 + ci) * 4 + kh) * 4 + kw];
      }
    }
  }
  s = s > 0.f ? s : 0.f;
  out[t] = (_Float16)s;
}

// ---------------------------------------------------------------------------
// Generic implicit-GEMM WMMA conv, C_in = C_out = 128, compile-time shape.
// One wave: 16 output pixels x 128 output channels.
// A frag: per-lane 2x b128 from NHWC input; B frag: 2x b128 from repacked weights.
// ---------------------------------------------------------------------------
template <int KH, int KW, int STRIDE, int PAD, int FLAGS>
__global__ __launch_bounds__(256) void conv_wmma_t(
    const _Float16* __restrict__ in, const _Float16* __restrict__ wt,
    const float* __restrict__ bias, const _Float16* __restrict__ res,
    _Float16* __restrict__ out,
    int Nn, int Hin, int Win, int Hout, int Wout) {
  constexpr int C = 128;
  const int lane = threadIdx.x & 31;
  const int wave = threadIdx.x >> 5;
  const int hi   = lane >> 4;     // half-wave selector
  const int col  = lane & 15;     // N column / A row (M)
  const long mbase = ((long)blockIdx.x * 8 + wave) * 16;

  // This lane's A-matrix row pixel (lanes 0-15 and 16-31 both map M=col).
  const long mrow = mbase + col;
  const int HW = Hout * Wout;
  const int n  = (int)(mrow / HW);
  const int rp = (int)(mrow % HW);
  const int oh = rp / Wout;
  const int ow = rp % Wout;

  v8f acc[8];
#pragma unroll
  for (int i = 0; i < 8; ++i) acc[i] = zero8f();

  constexpr int kArea = KH * KW;
#pragma unroll
  for (int kh = 0; kh < KH; ++kh) {
    const int ih = oh * STRIDE - PAD + kh;
    const bool vh = (unsigned)ih < (unsigned)Hin;
#pragma unroll
    for (int kw = 0; kw < KW; ++kw) {
      const int iw = ow * STRIDE - PAD + kw;
      const bool valid = vh && ((unsigned)iw < (unsigned)Win);
      const _Float16* src = in + (((long)n * Hin + ih) * Win + iw) * C;
#pragma unroll
      for (int cc = 0; cc < C; cc += 32) {
        v8h a0, a1;
        if (valid) {
          a0 = *(const v8h*)(src + cc + hi * 16);
          a1 = *(const v8h*)(src + cc + hi * 16 + 8);
        } else { a0 = zero8h(); a1 = zero8h(); }
        if (FLAGS & FLAG_RELU_IN) { a0 = relu8(a0); a1 = relu8(a1); }
        v16h A = CAT16(a0, a1);
        const long wBaseK = ((long)kh * KW + kw) * C + cc + hi * 16;
#pragma unroll
        for (int nb = 0; nb < 8; ++nb) {
          const _Float16* wp = wt + (long)(nb * 16 + col) * kArea * C + wBaseK;
          v8h b0 = *(const v8h*)(wp);
          v8h b1 = *(const v8h*)(wp + 8);
          v16h B = CAT16(b0, b1);
          acc[nb] = __builtin_amdgcn_wmma_f32_16x16x32_f16(
              false, A, false, B, (short)0, acc[nb], false, false);
        }
      }
    }
  }

#pragma unroll
  for (int nb = 0; nb < 8; ++nb) {
    const int oc = nb * 16 + col;
    const float bv = bias[oc];
#pragma unroll
    for (int r = 0; r < 8; ++r) {
      const long m = mbase + r + hi * 8;       // D: M = r + 8*hi, N = col
      float v = acc[nb][r] + bv;
      if (FLAGS & FLAG_ADD_RES) v += (float)res[m * C + oc];
      if (FLAGS & FLAG_RELU_OUT) v = v > 0.f ? v : 0.f;
      out[m * C + oc] = (_Float16)v;
    }
  }
}

// ---------------------------------------------------------------------------
// ConvTranspose2d 128->128, k4 s2 p1 via WMMA. Output pixels enumerated
// parity-major so each 16-pixel tile shares a uniform 4-tap set.
// m' = ((n*Hout + oh)*2 + owpar)*(Wout/2) + owhalf ; ow = 2*owhalf + owpar
// ---------------------------------------------------------------------------
__global__ __launch_bounds__(256) void convt_wmma_kernel(
    const _Float16* __restrict__ in, const _Float16* __restrict__ wt,
    const float* __restrict__ bias, _Float16* __restrict__ out,
    int Nn, int Hin, int Win, int Hout, int Wout) {
  constexpr int C = 128;
  const int lane = threadIdx.x & 31;
  const int wave = threadIdx.x >> 5;
  const int hi   = lane >> 4;
  const int col  = lane & 15;
  const long mbase = ((long)blockIdx.x * 8 + wave) * 16;
  const int Whalf = Wout >> 1;

  long mp = mbase + col;
  int owhalf = (int)(mp % Whalf); long tq = mp / Whalf;
  int owpar  = (int)(tq & 1);     tq >>= 1;
  int oh     = (int)(tq % Hout);
  int n      = (int)(tq / Hout);
  int ow     = owhalf * 2 + owpar;

  v8f acc[8];
#pragma unroll
  for (int i = 0; i < 8; ++i) acc[i] = zero8f();

#pragma unroll
  for (int khi = 0; khi < 2; ++khi) {
    const int kh = ((oh & 1) ^ 1) + 2 * khi;     // kh parity matches (oh+1)
    const int ih = (oh + 1 - kh) >> 1;
    const bool vh = (unsigned)ih < (unsigned)Hin;
#pragma unroll
    for (int kwi = 0; kwi < 2; ++kwi) {
      const int kw = ((ow & 1) ^ 1) + 2 * kwi;
      const int iw = (ow + 1 - kw) >> 1;
      const bool valid = vh && ((unsigned)iw < (unsigned)Win);
      const _Float16* src = in + (((long)n * Hin + ih) * Win + iw) * C;
#pragma unroll
      for (int cc = 0; cc < C; cc += 32) {
        v8h a0, a1;
        if (valid) {
          a0 = *(const v8h*)(src + cc + hi * 16);
          a1 = *(const v8h*)(src + cc + hi * 16 + 8);
        } else { a0 = zero8h(); a1 = zero8h(); }
        v16h A = CAT16(a0, a1);
        const long wBaseK = ((long)kh * 4 + kw) * C + cc + hi * 16;
#pragma unroll
        for (int nb = 0; nb < 8; ++nb) {
          const _Float16* wp = wt + (long)(nb * 16 + col) * (16 * C) + wBaseK;
          v8h b0 = *(const v8h*)(wp);
          v8h b1 = *(const v8h*)(wp + 8);
          v16h B = CAT16(b0, b1);
          acc[nb] = __builtin_amdgcn_wmma_f32_16x16x32_f16(
              false, A, false, B, (short)0, acc[nb], false, false);
        }
      }
    }
  }

#pragma unroll
  for (int nb = 0; nb < 8; ++nb) {
    const int oc = nb * 16 + col;
    const float bv = bias[oc];
#pragma unroll
    for (int r = 0; r < 8; ++r) {
      long mp2 = mbase + r + hi * 8;
      int owhalf2 = (int)(mp2 % Whalf); long t2 = mp2 / Whalf;
      int owpar2  = (int)(t2 & 1);      t2 >>= 1;
      int oh2     = (int)(t2 % Hout);
      int n2      = (int)(t2 / Hout);
      int ow2     = owhalf2 * 2 + owpar2;
      long pix = ((long)n2 * Hout + oh2) * Wout + ow2;
      float v = acc[nb][r] + bv;
      v = v > 0.f ? v : 0.f;                  // dec_tw1 output is relu'd
      out[pix * C + oc] = (_Float16)v;
    }
  }
}

// ---------------------------------------------------------------------------
// Final convT 128->3 (k4 s2 p1), VALU f32, NHWC f16 in -> NCHW f32 d_out.
// ---------------------------------------------------------------------------
__global__ __launch_bounds__(256) void convt2_valu_kernel(
    const _Float16* __restrict__ up, const float* __restrict__ w,
    const float* __restrict__ b, float* __restrict__ out) {
  long t = (long)blockIdx.x * 256 + threadIdx.x;     // 16*3*256*256 outputs
  int ow = (int)(t & 255); long r = t >> 8;
  int oh = (int)(r & 255); r >>= 8;
  int o  = (int)(r % 3);
  int n  = (int)(r / 3);
  float s = b[o];
#pragma unroll
  for (int khi = 0; khi < 2; ++khi) {
    int kh = ((oh & 1) ^ 1) + 2 * khi;
    int ih = (oh + 1 - kh) >> 1;
    if ((unsigned)ih >= 128u) continue;
#pragma unroll
    for (int kwi = 0; kwi < 2; ++kwi) {
      int kw = ((ow & 1) ^ 1) + 2 * kwi;
      int iw = (ow + 1 - kw) >> 1;
      if ((unsigned)iw >= 128u) continue;
      const _Float16* p = up + (((long)n * 128 + ih) * 128 + iw) * 128;
      const float* wc = w + (o * 16 + kh * 4 + kw);  // w[i][o][kh][kw], i-stride 48
      for (int c = 0; c < 128; ++c)
        s += (float)p[c] * wc[(long)c * 48];
    }
  }
  out[t] = s;
}

// ---------------------------------------------------------------------------
// VQ: per-pixel / per-code squared-norm helpers.
// ---------------------------------------------------------------------------
__global__ void zenorm_kernel(const _Float16* __restrict__ ze, float* __restrict__ zn) {
  long m = (long)blockIdx.x * 256 + threadIdx.x;     // 65536 pixels
  const _Float16* p = ze + m * 128;
  float s = 0.f;
  for (int c = 0; c < 128; ++c) { float v = (float)p[c]; s += v * v; }
  zn[m] = s;
}
__global__ void embnorm_kernel(const _Float16* __restrict__ e, float* __restrict__ en) {
  int k = blockIdx.x * 256 + threadIdx.x;
  if (k >= 512) return;
  const _Float16* p = e + (long)k * 128;
  float s = 0.f;
  for (int c = 0; c < 128; ++c) { float v = (float)p[c]; s += v * v; }
  en[k] = s;
}

// ---------------------------------------------------------------------------
// VQ argmin: WMMA ze(16px x 128) x embT(128 x 16 codes), 32 code blocks,
// d2 = |ze|^2 + |e|^2 - 2 ze.e ; cross-lane argmin via LDS.
// ---------------------------------------------------------------------------
__global__ __launch_bounds__(256) void vq_wmma_kernel(
    const _Float16* __restrict__ ze, const _Float16* __restrict__ embh,
    const float* __restrict__ zeN, const float* __restrict__ embN,
    unsigned* __restrict__ idxOut) {
  constexpr int C = 128;
  __shared__ float sB[8][16][16];
  __shared__ int   sI[8][16][16];
  const int lane = threadIdx.x & 31;
  const int wave = threadIdx.x >> 5;
  const int hi   = lane >> 4;
  const int col  = lane & 15;
  const long mbase = ((long)blockIdx.x * 8 + wave) * 16;

  const _Float16* src = ze + (mbase + col) * C;
  float zn[8];
#pragma unroll
  for (int r = 0; r < 8; ++r) zn[r] = zeN[mbase + r + hi * 8];

  float best[8]; int bidx[8];
#pragma unroll
  for (int r = 0; r < 8; ++r) { best[r] = 3.4e38f; bidx[r] = 0; }

  for (int nb = 0; nb < 32; ++nb) {
    v8f acc = zero8f();
#pragma unroll
    for (int cc = 0; cc < C; cc += 32) {
      v8h a0 = *(const v8h*)(src + cc + hi * 16);
      v8h a1 = *(const v8h*)(src + cc + hi * 16 + 8);
      v16h A = CAT16(a0, a1);
      const _Float16* wp = embh + (long)(nb * 16 + col) * C + cc + hi * 16;
      v8h b0 = *(const v8h*)(wp);
      v8h b1 = *(const v8h*)(wp + 8);
      v16h B = CAT16(b0, b1);
      acc = __builtin_amdgcn_wmma_f32_16x16x32_f16(
          false, A, false, B, (short)0, acc, false, false);
    }
    float en = embN[nb * 16 + col];
#pragma unroll
    for (int r = 0; r < 8; ++r) {
      float d = zn[r] + en - 2.f * acc[r];
      if (d < best[r]) { best[r] = d; bidx[r] = nb * 16 + col; }
    }
  }
#pragma unroll
  for (int r = 0; r < 8; ++r) {
    sB[wave][r + 8 * hi][col] = best[r];
    sI[wave][r + 8 * hi][col] = bidx[r];
  }
  __syncthreads();
  if (lane < 16) {
    float bb = sB[wave][lane][0]; int bi = sI[wave][lane][0];
#pragma unroll
    for (int c = 1; c < 16; ++c)
      if (sB[wave][lane][c] < bb) { bb = sB[wave][lane][c]; bi = sI[wave][lane][c]; }
    idxOut[mbase + lane] = (unsigned)bi;
  }
}

// zq = emb[idx]: build decoder input (f16 NHWC) + zq f32 NCHW output.
__global__ void zq_build_kernel(const unsigned* __restrict__ idx, const float* __restrict__ emb,
                                _Float16* __restrict__ decin, float* __restrict__ zqOut) {
  long t = (long)blockIdx.x * 256 + threadIdx.x;     // 65536*128
  int c = (int)(t & 127); long m = t >> 7;
  unsigned i = idx[m];
  float v = emb[(long)i * 128 + c];
  decin[t] = (_Float16)v;
  int w = (int)(m & 63), h = (int)((m >> 6) & 63), n = (int)(m >> 12);
  zqOut[(((long)n * 128 + c) * 64 + h) * 64 + w] = v;
}

// ze f16 NHWC (64x64) -> f32 NCHW output region.
__global__ void ze_out_kernel(const _Float16* __restrict__ a, float* __restrict__ o) {
  long t = (long)blockIdx.x * 256 + threadIdx.x;
  int c = (int)(t & 127); long m = t >> 7;
  int w = (int)(m & 63), h = (int)((m >> 6) & 63), n = (int)(m >> 12);
  o[(((long)n * 128 + c) * 64 + h) * 64 + w] = (float)a[t];
}

// ---------------------------------------------------------------------------
extern "C" void kernel_launch(void* const* d_in, const int* in_sizes, int n_in,
                              void* d_out, int out_size, void* d_ws, size_t ws_size,
                              hipStream_t stream) {
  (void)in_sizes; (void)n_in; (void)out_size; (void)ws_size;
  // --- inputs (setup_inputs dict order) ---
  const float* x       = (const float*)d_in[0];
  const float* emb     = (const float*)d_in[1];
  const float* enc_w1  = (const float*)d_in[2];
  const float* enc_b1  = (const float*)d_in[3];
  const float* enc_w2  = (const float*)d_in[4];
  const float* enc_b2  = (const float*)d_in[5];
  const float* enc_w3  = (const float*)d_in[6];
  const float* enc_b3  = (const float*)d_in[7];
  const float* er1_w1  = (const float*)d_in[8];
  const float* er1_b1  = (const float*)d_in[9];
  const float* er1_w2  = (const float*)d_in[10];
  const float* er1_b2  = (const float*)d_in[11];
  const float* er2_w1  = (const float*)d_in[12];
  const float* er2_b1  = (const float*)d_in[13];
  const float* er2_w2  = (const float*)d_in[14];
  const float* er2_b2  = (const float*)d_in[15];
  const float* dr1_w1  = (const float*)d_in[16];
  const float* dr1_b1  = (const float*)d_in[17];
  const float* dr1_w2  = (const float*)d_in[18];
  const float* dr1_b2  = (const float*)d_in[19];
  const float* dr2_w1  = (const float*)d_in[20];
  const float* dr2_b1  = (const float*)d_in[21];
  const float* dr2_w2  = (const float*)d_in[22];
  const float* dr2_b2  = (const float*)d_in[23];
  const float* dec_w1  = (const float*)d_in[24];
  const float* dec_b1  = (const float*)d_in[25];
  const float* dec_tw1 = (const float*)d_in[26];
  const float* dec_tb1 = (const float*)d_in[27];
  const float* dec_tw2 = (const float*)d_in[28];
  const float* dec_tb2 = (const float*)d_in[29];

  float* xhat_out = (float*)d_out;                       // 16*3*256*256
  float* ze_out_p = xhat_out + 3145728;                  // 16*128*64*64
  float* zq_out_p = ze_out_p + 8388608;

  // --- workspace carve-out ---
  char* ws = (char*)d_ws;
  size_t off = 0;
  auto alloc = [&](size_t bytes) -> void* {
    void* p = (void*)(ws + off);
    off += (bytes + 255) & ~(size_t)255;
    return p;
  };
  _Float16* WT_ENC2 = (_Float16*)alloc(262144 * 2);
  _Float16* WT_ENC3 = (_Float16*)alloc(147456 * 2);
  _Float16* WT_ER1A = (_Float16*)alloc(147456 * 2);
  _Float16* WT_ER1B = (_Float16*)alloc(16384 * 2);
  _Float16* WT_ER2A = (_Float16*)alloc(147456 * 2);
  _Float16* WT_ER2B = (_Float16*)alloc(16384 * 2);
  _Float16* WT_DEC1 = (_Float16*)alloc(147456 * 2);
  _Float16* WT_DR1A = (_Float16*)alloc(147456 * 2);
  _Float16* WT_DR1B = (_Float16*)alloc(16384 * 2);
  _Float16* WT_DR2A = (_Float16*)alloc(147456 * 2);
  _Float16* WT_DR2B = (_Float16*)alloc(16384 * 2);
  _Float16* WT_T1   = (_Float16*)alloc(262144 * 2);
  _Float16* EMBH    = (_Float16*)alloc(65536 * 2);
  float*    EMBN    = (float*)alloc(512 * 4);
  float*    ZEN     = (float*)alloc(65536 * 4);
  unsigned* IDX     = (unsigned*)alloc(65536 * 4);
  _Float16* A1      = (_Float16*)alloc((size_t)33554432 * 2);  // 16x128x128x128 (also up1)
  _Float16* B0      = (_Float16*)alloc((size_t)8388608 * 2);   // 16x64x64x128
  _Float16* B1      = (_Float16*)alloc((size_t)8388608 * 2);
  _Float16* B2      = (_Float16*)alloc((size_t)8388608 * 2);
  _Float16* ZEBUF   = (_Float16*)alloc((size_t)8388608 * 2);

  // --- 0. weight repack + embedding convert ---
  repack_w_kernel <<<1024, 256, 0, stream>>>(enc_w2, WT_ENC2, 128, 128, 4, 4);
  repack_w_kernel <<<576,  256, 0, stream>>>(enc_w3, WT_ENC3, 128, 128, 3, 3);
  repack_w_kernel <<<576,  256, 0, stream>>>(er1_w1, WT_ER1A, 128, 128, 3, 3);
  repack_w_kernel <<<64,   256, 0, stream>>>(er1_w2, WT_ER1B, 128, 128, 1, 1);
  repack_w_kernel <<<576,  256, 0, stream>>>(er2_w1, WT_ER2A, 128, 128, 3, 3);
  repack_w_kernel <<<64,   256, 0, stream>>>(er2_w2, WT_ER2B, 128, 128, 1, 1);
  repack_w_kernel <<<576,  256, 0, stream>>>(dec_w1, WT_DEC1, 128, 128, 3, 3);
  repack_w_kernel <<<576,  256, 0, stream>>>(dr1_w1, WT_DR1A, 128, 128, 3, 3);
  repack_w_kernel <<<64,   256, 0, stream>>>(dr1_w2, WT_DR1B, 128, 128, 1, 1);
  repack_w_kernel <<<576,  256, 0, stream>>>(dr2_w1, WT_DR2A, 128, 128, 3, 3);
  repack_w_kernel <<<64,   256, 0, stream>>>(dr2_w2, WT_DR2B, 128, 128, 1, 1);
  repack_wt_kernel<<<1024, 256, 0, stream>>>(dec_tw1, WT_T1, 128, 128, 4, 4);
  cvt_f32_to_f16_kernel<<<256, 256, 0, stream>>>(emb, EMBH, 65536);
  embnorm_kernel<<<2, 256, 0, stream>>>(EMBH, EMBN);

  // --- 1. encoder ---
  conv1_valu_kernel<<<131072, 256, 0, stream>>>(x, enc_w1, enc_b1, A1);
  // enc conv2: 128x128 -> 64x64, k4 s2 p1, relu
  conv_wmma_t<4, 4, 2, 1, FLAG_RELU_OUT><<<512, 256, 0, stream>>>(
      A1, WT_ENC2, enc_b2, nullptr, B0, 16, 128, 128, 64, 64);
  // enc conv3: 3x3 s1 p1
  conv_wmma_t<3, 3, 1, 1, 0><<<512, 256, 0, stream>>>(
      B0, WT_ENC3, enc_b3, nullptr, B1, 16, 64, 64, 64, 64);
  // res er1
  conv_wmma_t<3, 3, 1, 1, FLAG_RELU_IN><<<512, 256, 0, stream>>>(
      B1, WT_ER1A, er1_b1, nullptr, B2, 16, 64, 64, 64, 64);
  conv_wmma_t<1, 1, 1, 0, FLAG_RELU_IN | FLAG_ADD_RES><<<512, 256, 0, stream>>>(
      B2, WT_ER1B, er1_b2, B1, B0, 16, 64, 64, 64, 64);
  // res er2 -> ze
  conv_wmma_t<3, 3, 1, 1, FLAG_RELU_IN><<<512, 256, 0, stream>>>(
      B0, WT_ER2A, er2_b1, nullptr, B2, 16, 64, 64, 64, 64);
  conv_wmma_t<1, 1, 1, 0, FLAG_RELU_IN | FLAG_ADD_RES><<<512, 256, 0, stream>>>(
      B2, WT_ER2B, er2_b2, B0, ZEBUF, 16, 64, 64, 64, 64);

  // --- 2. VQ ---
  zenorm_kernel<<<256, 256, 0, stream>>>(ZEBUF, ZEN);
  vq_wmma_kernel<<<512, 256, 0, stream>>>(ZEBUF, EMBH, ZEN, EMBN, IDX);
  zq_build_kernel<<<32768, 256, 0, stream>>>(IDX, emb, B1, zq_out_p);  // decoder input = zq
  ze_out_kernel<<<32768, 256, 0, stream>>>(ZEBUF, ze_out_p);

  // --- 3. decoder ---
  conv_wmma_t<3, 3, 1, 1, 0><<<512, 256, 0, stream>>>(
      B1, WT_DEC1, dec_b1, nullptr, B0, 16, 64, 64, 64, 64);
  // res dr1
  conv_wmma_t<3, 3, 1, 1, FLAG_RELU_IN><<<512, 256, 0, stream>>>(
      B0, WT_DR1A, dr1_b1, nullptr, B2, 16, 64, 64, 64, 64);
  conv_wmma_t<1, 1, 1, 0, FLAG_RELU_IN | FLAG_ADD_RES><<<512, 256, 0, stream>>>(
      B2, WT_DR1B, dr1_b2, B0, B1, 16, 64, 64, 64, 64);
  // res dr2
  conv_wmma_t<3, 3, 1, 1, FLAG_RELU_IN><<<512, 256, 0, stream>>>(
      B1, WT_DR2A, dr2_b1, nullptr, B2, 16, 64, 64, 64, 64);
  conv_wmma_t<1, 1, 1, 0, FLAG_RELU_IN | FLAG_ADD_RES><<<512, 256, 0, stream>>>(
      B2, WT_DR2B, dr2_b2, B1, B0, 16, 64, 64, 64, 64);
  // convT1: 64x64 -> 128x128, relu (A1 reused as up1)
  convt_wmma_kernel<<<2048, 256, 0, stream>>>(B0, WT_T1, dec_tb1, A1,
      16, 64, 64, 128, 128);
  // convT2: 128x128 -> 256x256, 3 channels, f32 NCHW out
  convt2_valu_kernel<<<12288, 256, 0, stream>>>(A1, dec_tw2, dec_tb2, xhat_out);
}
